// MultiGraphClassifier_32375463477756
// MI455X (gfx1250) — matrix operand order repaired and checked by reading, not compiled, not run on hardware
//
#include <hip/hip_runtime.h>
#include <hip/hip_bf16.h>
#include <math.h>

// ---------------------------------------------------------------------------
// MultiGraphClassifier on gfx1250 (MI455X, wave32, WMMA).
// Dense node GEMMs: v_wmma_f32_16x16x32_f16, f16 operands staged contiguously
// so every fragment is 2x global_load_b128. Edge aggregation: coalesced b128
// loads + fp32 global atomics (the memory-bound part of the model).
// ---------------------------------------------------------------------------

typedef __attribute__((ext_vector_type(16))) _Float16 v16h;
typedef __attribute__((ext_vector_type(8)))  _Float16 v8h;
typedef __attribute__((ext_vector_type(8)))  float    v8f;

#define NA 50000
#define NF 100000
#define EA 800000
#define EF 1600000
#define DA 256
#define DF 128
#define HID 128
#define UNI 64
#define NCLS 10
#define FUSION 0.1f

// ------------------------------ utility kernels ----------------------------

__global__ void fill_f32(float* __restrict__ p, float v, int n) {
  int i = blockIdx.x * blockDim.x + threadIdx.x;
  if (i < n) p[i] = v;
}

__global__ void degree_kernel(const int* __restrict__ src, const int* __restrict__ dst,
                              int E, float* __restrict__ dout, float* __restrict__ din) {
  int e = blockIdx.x * blockDim.x + threadIdx.x;
  if (e < E) {
    atomicAdd(&dout[src[e]], 1.0f);
    atomicAdd(&din[dst[e]], 1.0f);
  }
}

// d <- (max(d,1))^{-1/2}   (matches clip(deg,1)**-0.5)
__global__ void rsqrt_clamp_kernel(float* __restrict__ d, int n) {
  int i = blockIdx.x * blockDim.x + threadIdx.x;
  if (i < n) d[i] = rsqrtf(fmaxf(d[i], 1.0f));
}

// fp32 -> f16, 4 elems/thread (n % 4 == 0)
__global__ void cvt_f32_to_f16(const float* __restrict__ in, _Float16* __restrict__ out,
                               int n4) {
  int i = blockIdx.x * blockDim.x + threadIdx.x;
  if (i < n4) {
    float4 v = ((const float4*)in)[i];
    out[i * 4 + 0] = (_Float16)v.x;
    out[i * 4 + 1] = (_Float16)v.y;
    out[i * 4 + 2] = (_Float16)v.z;
    out[i * 4 + 3] = (_Float16)v.w;
  }
}

// W[K,N] fp32 -> Wt[N,K] f16 (weights are tiny)
__global__ void transpose_w_f16(const float* __restrict__ Wm, _Float16* __restrict__ Wt,
                                int K, int N) {
  int idx = blockIdx.x * blockDim.x + threadIdx.x;
  if (idx < K * N) {
    int k = idx / N, n = idx % N;
    Wt[(size_t)n * K + k] = (_Float16)Wm[idx];
  }
}

// ------------------------------ WMMA GEMM ----------------------------------
// C[M,N] = (HAS_SCALE ? diag(rowscale) : I) * (A[M,K] @ B[K,N]) (+ bias)
// A: f16 row-major [M,K]; Bt: f16 [N,K] (transposed so fragments are linear).
// One wave -> 16x64 tile (A fragment reused by 4 WMMAs); 4 waves/block -> 64x64.

__device__ inline v16h load_a_frag(const _Float16* __restrict__ A, int K,
                                   int m, int k0, int half) {
  // 16-bit A 16x32 (ISA 7.12.2): lane half selects +8 on K;
  // elems 0..7 -> K [k0+8h, +8), elems 8..15 -> K [k0+16+8h, +8): 2x b128.
  const _Float16* row = A + (size_t)m * K + k0 + 8 * half;
  v8h lo = *(const v8h*)row;
  v8h hi = *(const v8h*)(row + 16);
  v16h a;
#pragma unroll
  for (int i = 0; i < 8; ++i) { a[i] = lo[i]; a[i + 8] = hi[i]; }
  return a;
}

__device__ inline v16h load_b_frag(const _Float16* __restrict__ Bt, int K,
                                   int n, int k0, int half) {
  // 16-bit B 32x16: same K order as A: elems 0..7 -> K [k0+8h, +8), elems 8..15 -> K [k0+16+8h, +8): 2x b128.
  const _Float16* row = Bt + (size_t)n * K + k0 + 8 * half;
  v8h lo = *(const v8h*)row;
  v8h hi = *(const v8h*)(row + 16);
  v16h b;
#pragma unroll
  for (int i = 0; i < 8; ++i) { b[i] = lo[i]; b[i + 8] = hi[i]; }
  return b;
}

template <bool HAS_BIAS, bool HAS_SCALE, bool WRITE_F32, bool WRITE_F16>
__global__ void gemm_wmma(const _Float16* __restrict__ A, const _Float16* __restrict__ Bt,
                          const float* __restrict__ bias, const float* __restrict__ rowscale,
                          float* __restrict__ C, _Float16* __restrict__ C16,
                          int M, int K, int N) {
  const int lane = threadIdx.x & 31;
  const int wave = threadIdx.x >> 5;                 // 0..3 -> M sub-tile
  const int m0 = (blockIdx.x * 4 + wave) * 16;
  const int n0 = blockIdx.y * 64;
  if (m0 >= M) return;                               // wave-uniform: EXEC stays all-1s

  const int half = lane >> 4;
  const int ml   = m0 + (lane & 15);                 // A row for this lane
  const int nl   = lane & 15;

  v8f c0 = {}, c1 = {}, c2 = {}, c3 = {};
  for (int k0 = 0; k0 < K; k0 += 32) {
    v16h a = load_a_frag(A, K, ml, k0, half);
    v16h b0 = load_b_frag(Bt, K, n0 + nl,      k0, half);
    v16h b1 = load_b_frag(Bt, K, n0 + 16 + nl, k0, half);
    v16h b2 = load_b_frag(Bt, K, n0 + 32 + nl, k0, half);
    v16h b3 = load_b_frag(Bt, K, n0 + 48 + nl, k0, half);
    c0 = __builtin_amdgcn_wmma_f32_16x16x32_f16(false, a, false, b0, (short)0, c0, false, false);
    c1 = __builtin_amdgcn_wmma_f32_16x16x32_f16(false, a, false, b1, (short)0, c1, false, false);
    c2 = __builtin_amdgcn_wmma_f32_16x16x32_f16(false, a, false, b2, (short)0, c2, false, false);
    c3 = __builtin_amdgcn_wmma_f32_16x16x32_f16(false, a, false, b3, (short)0, c3, false, false);
  }

  // C/D layout: VGPR r holds M = r + 8*half, N = lane&15 (+16 per sub-tile).
  v8f acc[4] = {c0, c1, c2, c3};
#pragma unroll
  for (int t = 0; t < 4; ++t) {
    int n = n0 + 16 * t + nl;
    float bn = HAS_BIAS ? bias[n] : 0.0f;
#pragma unroll
    for (int r = 0; r < 8; ++r) {
      int m = m0 + r + 8 * half;
      float v = acc[t][r];
      if (HAS_SCALE) v *= rowscale[m];
      v += bn;
      if (WRITE_F32) C[(size_t)m * N + n] = v;
      if (WRITE_F16) C16[(size_t)m * N + n] = (_Float16)v;
    }
  }
}

// ------------------------------ edge scatter -------------------------------
// agg[dst[e], :] += h[src[e], :]  -- b128 load per lane, 4 fp32 atomics.

__global__ void edge_scatter(const int* __restrict__ src, const int* __restrict__ dst,
                             const float* __restrict__ h, float* __restrict__ agg, int E) {
  int gid  = blockIdx.x * blockDim.x + threadIdx.x;
  int e    = gid >> 5;
  int lane = gid & 31;
  if (e >= E) return;
  const float4* hs = (const float4*)(h + (size_t)src[e] * HID);
  float* ad = agg + (size_t)dst[e] * HID + lane * 4;
  float4 v = hs[lane];
  atomicAdd(ad + 0, v.x);
  atomicAdd(ad + 1, v.y);
  atomicAdd(ad + 2, v.z);
  atomicAdd(ad + 3, v.w);
}

// out = relu(agg * deg_in^{-1/2} + bias); also emit f16 copy for the next GEMM
__global__ void conv_epilogue(const float* __restrict__ agg, const float* __restrict__ rs_in,
                              const float* __restrict__ bias, float* __restrict__ out,
                              _Float16* __restrict__ out16, int M, int N) {
  size_t idx = (size_t)blockIdx.x * blockDim.x + threadIdx.x;
  if (idx >= (size_t)M * N) return;
  int i = (int)(idx / N), j = (int)(idx % N);
  float v = fmaxf(agg[idx] * rs_in[i] + bias[j], 0.0f);
  out[idx] = v;
  out16[idx] = (_Float16)v;
}

// ------------------------------ reductions ---------------------------------

__global__ void colsum_kernel(const float* __restrict__ X, int M, int N,
                              float* __restrict__ out) {
  int col = threadIdx.x;                             // blockDim.x == N
  float acc = 0.0f;
  for (int r = blockIdx.x; r < M; r += gridDim.x) acc += X[(size_t)r * N + col];
  atomicAdd(&out[col], acc);
}

// column max of a ReLU output (>= 0) via ordered int-bit trick
__global__ void colmax_kernel(const float* __restrict__ X, int M, int N,
                              int* __restrict__ out) {
  int col = threadIdx.x;
  float acc = 0.0f;
  for (int r = blockIdx.x; r < M; r += gridDim.x) acc = fmaxf(acc, X[(size_t)r * N + col]);
  atomicMax(&out[col], __float_as_int(acc));
}

// X16[:, col] = (f16)(X[:, col] + FUSION * S[col])  (post-fusion f16 staging)
__global__ void fuse_add_f16(const float* __restrict__ X, const float* __restrict__ S,
                             _Float16* __restrict__ X16, int M, int N) {
  size_t idx = (size_t)blockIdx.x * blockDim.x + threadIdx.x;
  if (idx >= (size_t)M * N) return;
  X16[idx] = (_Float16)(X[idx] + FUSION * S[idx % N]);
}

// ------------------------------ finalize -----------------------------------
// apig_emb = colsum_a / NA (mean nodes), fcg_emb = colmax_f (max nodes).
// norm_embed each; softmax over a single logit == 1.0 exactly, so the
// attention weights drop out: emb = norm(a) + norm(f). Then emb @ W_cls + b.

__global__ void finalize_kernel(const float* __restrict__ col_sum_a,
                                const float* __restrict__ col_max_f,
                                const float* __restrict__ W_cls,
                                const float* __restrict__ b_cls,
                                float* __restrict__ out) {
  __shared__ float a[HID], f[HID], red[HID];
  const int t = threadIdx.x;                         // 128 threads
  a[t] = col_sum_a[t] * (1.0f / (float)NA);
  f[t] = col_max_f[t];
  __syncthreads();

  for (int w = 0; w < 2; ++w) {
    float* e = (w == 0) ? a : f;
    // mean
    red[t] = e[t]; __syncthreads();
    for (int s = HID / 2; s > 0; s >>= 1) { if (t < s) red[t] += red[t + s]; __syncthreads(); }
    float mean = red[0] * (1.0f / (float)HID); __syncthreads();
    // unbiased std (ddof=1, matches torch.std)
    float d = e[t] - mean;
    red[t] = d * d; __syncthreads();
    for (int s = HID / 2; s > 0; s >>= 1) { if (t < s) red[t] += red[t + s]; __syncthreads(); }
    float inv_sd = rsqrtf(red[0] * (1.0f / (float)(HID - 1))); __syncthreads();
    float z = d * inv_sd;
    // min
    red[t] = z; __syncthreads();
    for (int s = HID / 2; s > 0; s >>= 1) { if (t < s) red[t] = fminf(red[t], red[t + s]); __syncthreads(); }
    float mn = red[0]; __syncthreads();
    // max
    red[t] = z; __syncthreads();
    for (int s = HID / 2; s > 0; s >>= 1) { if (t < s) red[t] = fmaxf(red[t], red[t + s]); __syncthreads(); }
    float mx = red[0]; __syncthreads();
    e[t] = (z - mn) / (mx - mn);
    __syncthreads();
  }

  red[t] = a[t] + f[t];                              // wa == wf == 1.0 exactly
  __syncthreads();
  if (t < NCLS) {
    float acc = b_cls[t];
    for (int j = 0; j < HID; ++j) acc += red[j] * W_cls[j * NCLS + t];
    out[t] = acc;
  }
}

// ------------------------------ launcher -----------------------------------

extern "C" void kernel_launch(void* const* d_in, const int* in_sizes, int n_in,
                              void* d_out, int out_size, void* d_ws, size_t ws_size,
                              hipStream_t stream) {
  (void)in_sizes; (void)n_in; (void)out_size; (void)ws_size;

  const float* apig_feat = (const float*)d_in[0];
  const float* fcg_feat  = (const float*)d_in[1];
  const float* W_a1 = (const float*)d_in[2];
  const float* b_a1 = (const float*)d_in[3];
  const float* W_a2 = (const float*)d_in[4];
  const float* b_a2 = (const float*)d_in[5];
  const float* W_f1 = (const float*)d_in[6];
  const float* b_f1 = (const float*)d_in[7];
  const float* W_f2 = (const float*)d_in[8];
  const float* b_f2 = (const float*)d_in[9];
  const float* W1   = (const float*)d_in[10];
  const float* b1   = (const float*)d_in[11];
  const float* W2   = (const float*)d_in[12];
  const float* b2   = (const float*)d_in[13];
  // d_in[14]=W_attn, d_in[15]=b_attn: mathematically eliminated (singleton softmax)
  const float* W_cls = (const float*)d_in[16];
  const float* b_cls = (const float*)d_in[17];
  const int* apig_src = (const int*)d_in[18];
  const int* apig_dst = (const int*)d_in[19];
  const int* fcg_src  = (const int*)d_in[20];
  const int* fcg_dst  = (const int*)d_in[21];

  // ---- workspace carve-up (256B-aligned slabs) ----
  float* Wsp = (float*)d_ws;
  size_t off = 0;
  auto allocF = [&](size_t n) { float* p = Wsp + off; off += (n + 63) & ~(size_t)63; return p; };
  auto allocH = [&](size_t n) { return (_Float16*)allocF((n + 1) / 2); };

  float* rs_out_a = allocF(NA);
  float* rs_in_a  = allocF(NA);
  float* rs_out_f = allocF(NF);
  float* rs_in_f  = allocF(NF);
  float* xw_a  = allocF((size_t)NA * HID);   // GEMM out, reused as conv output h
  float* agg_a = allocF((size_t)NA * HID);
  float* xw_f  = allocF((size_t)NF * HID);
  float* agg_f = allocF((size_t)NF * HID);
  float* enc_a = allocF((size_t)NA * UNI);
  float* enc_f = allocF((size_t)NF * UNI);
  float* sum_a = allocF(UNI);
  float* sum_f = allocF(UNI);
  float* col_a = allocF(HID);                // column sums of h_a2
  float* col_f = allocF(HID);                // column max of h_f2 (int bits)

  _Float16* fa16    = allocH((size_t)NA * DA);   // apig features f16
  _Float16* ff16    = allocH((size_t)NF * DF);   // fcg features f16
  _Float16* h16_a   = allocH((size_t)NA * HID);  // conv outputs / dec outputs f16
  _Float16* h16_f   = allocH((size_t)NF * HID);
  _Float16* enc16_a = allocH((size_t)NA * UNI);
  _Float16* enc16_f = allocH((size_t)NF * UNI);
  _Float16* dec16_a = allocH((size_t)NA * HID);
  _Float16* dec16_f = allocH((size_t)NF * HID);
  _Float16* wt_a1 = allocH((size_t)DA * HID);    // Wt: f16 [N,K]
  _Float16* wt_f1 = allocH((size_t)DF * HID);
  _Float16* wt_a2 = allocH((size_t)HID * HID);
  _Float16* wt_f2 = allocH((size_t)HID * HID);
  _Float16* wt1   = allocH((size_t)HID * UNI);
  _Float16* wt2   = allocH((size_t)UNI * HID);

  const int TPB = 256;
  auto blks = [](size_t n, int tpb) { return (int)((n + tpb - 1) / tpb); };
  auto gemm_grid = [](int M, int N) { return dim3((M / 16 + 3) / 4, N / 64); };

  // ---- degrees -> rsqrt scales ----
  fill_f32<<<blks(NA, TPB), TPB, 0, stream>>>(rs_out_a, 0.0f, NA);
  fill_f32<<<blks(NA, TPB), TPB, 0, stream>>>(rs_in_a, 0.0f, NA);
  fill_f32<<<blks(NF, TPB), TPB, 0, stream>>>(rs_out_f, 0.0f, NF);
  fill_f32<<<blks(NF, TPB), TPB, 0, stream>>>(rs_in_f, 0.0f, NF);
  degree_kernel<<<blks(EA, TPB), TPB, 0, stream>>>(apig_src, apig_dst, EA, rs_out_a, rs_in_a);
  degree_kernel<<<blks(EF, TPB), TPB, 0, stream>>>(fcg_src, fcg_dst, EF, rs_out_f, rs_in_f);
  rsqrt_clamp_kernel<<<blks(NA, TPB), TPB, 0, stream>>>(rs_out_a, NA);
  rsqrt_clamp_kernel<<<blks(NA, TPB), TPB, 0, stream>>>(rs_in_a, NA);
  rsqrt_clamp_kernel<<<blks(NF, TPB), TPB, 0, stream>>>(rs_out_f, NF);
  rsqrt_clamp_kernel<<<blks(NF, TPB), TPB, 0, stream>>>(rs_in_f, NF);

  // ---- f16 staging: features + transposed weights ----
  cvt_f32_to_f16<<<blks((size_t)NA * DA / 4, TPB), TPB, 0, stream>>>(apig_feat, fa16, NA * DA / 4);
  cvt_f32_to_f16<<<blks((size_t)NF * DF / 4, TPB), TPB, 0, stream>>>(fcg_feat, ff16, NF * DF / 4);
  transpose_w_f16<<<blks(DA * HID, TPB), TPB, 0, stream>>>(W_a1, wt_a1, DA, HID);
  transpose_w_f16<<<blks(DF * HID, TPB), TPB, 0, stream>>>(W_f1, wt_f1, DF, HID);
  transpose_w_f16<<<blks(HID * HID, TPB), TPB, 0, stream>>>(W_a2, wt_a2, HID, HID);
  transpose_w_f16<<<blks(HID * HID, TPB), TPB, 0, stream>>>(W_f2, wt_f2, HID, HID);
  transpose_w_f16<<<blks(HID * UNI, TPB), TPB, 0, stream>>>(W1, wt1, HID, UNI);
  transpose_w_f16<<<blks(UNI * HID, TPB), TPB, 0, stream>>>(W2, wt2, UNI, HID);

  // ---- conv1 (A): xw = diag(rs_out)*(x@W_a1); scatter; epilogue -> h in xw_a/h16_a
  gemm_wmma<false, true, true, false><<<gemm_grid(NA, HID), 128, 0, stream>>>(
      fa16, wt_a1, nullptr, rs_out_a, xw_a, nullptr, NA, DA, HID);
  fill_f32<<<blks((size_t)NA * HID, TPB), TPB, 0, stream>>>(agg_a, 0.0f, NA * HID);
  edge_scatter<<<blks((size_t)EA * 32, TPB), TPB, 0, stream>>>(apig_src, apig_dst, xw_a, agg_a, EA);
  conv_epilogue<<<blks((size_t)NA * HID, TPB), TPB, 0, stream>>>(agg_a, rs_in_a, b_a1, xw_a, h16_a, NA, HID);

  // ---- conv1 (F)
  gemm_wmma<false, true, true, false><<<gemm_grid(NF, HID), 128, 0, stream>>>(
      ff16, wt_f1, nullptr, rs_out_f, xw_f, nullptr, NF, DF, HID);
  fill_f32<<<blks((size_t)NF * HID, TPB), TPB, 0, stream>>>(agg_f, 0.0f, NF * HID);
  edge_scatter<<<blks((size_t)EF * 32, TPB), TPB, 0, stream>>>(fcg_src, fcg_dst, xw_f, agg_f, EF);
  conv_epilogue<<<blks((size_t)NF * HID, TPB), TPB, 0, stream>>>(agg_f, rs_in_f, b_f1, xw_f, h16_f, NF, HID);

  // ---- encoder: enc = h1 @ W1 + b1  (fp32 out for colsum + fusion)
  gemm_wmma<true, false, true, false><<<gemm_grid(NA, UNI), 128, 0, stream>>>(
      h16_a, wt1, b1, nullptr, enc_a, nullptr, NA, HID, UNI);
  gemm_wmma<true, false, true, false><<<gemm_grid(NF, UNI), 128, 0, stream>>>(
      h16_f, wt1, b1, nullptr, enc_f, nullptr, NF, HID, UNI);

  // ---- cross-graph fusion (sums are pre-fusion, per reference) ----
  fill_f32<<<1, UNI, 0, stream>>>(sum_a, 0.0f, UNI);
  fill_f32<<<1, UNI, 0, stream>>>(sum_f, 0.0f, UNI);
  colsum_kernel<<<512, UNI, 0, stream>>>(enc_a, NA, UNI, sum_a);
  colsum_kernel<<<512, UNI, 0, stream>>>(enc_f, NF, UNI, sum_f);
  fuse_add_f16<<<blks((size_t)NA * UNI, TPB), TPB, 0, stream>>>(enc_a, sum_f, enc16_a, NA, UNI);
  fuse_add_f16<<<blks((size_t)NF * UNI, TPB), TPB, 0, stream>>>(enc_f, sum_a, enc16_f, NF, UNI);

  // ---- decoder: dec = enc @ W2 + b2 (f16-only out: feeds conv2 GEMM) ----
  gemm_wmma<true, false, false, true><<<gemm_grid(NA, HID), 128, 0, stream>>>(
      enc16_a, wt2, b2, nullptr, nullptr, dec16_a, NA, UNI, HID);
  gemm_wmma<true, false, false, true><<<gemm_grid(NF, HID), 128, 0, stream>>>(
      enc16_f, wt2, b2, nullptr, nullptr, dec16_f, NF, UNI, HID);

  // ---- conv2 (A): h_a2 in xw_a ----
  gemm_wmma<false, true, true, false><<<gemm_grid(NA, HID), 128, 0, stream>>>(
      dec16_a, wt_a2, nullptr, rs_out_a, xw_a, nullptr, NA, HID, HID);
  fill_f32<<<blks((size_t)NA * HID, TPB), TPB, 0, stream>>>(agg_a, 0.0f, NA * HID);
  edge_scatter<<<blks((size_t)EA * 32, TPB), TPB, 0, stream>>>(apig_src, apig_dst, xw_a, agg_a, EA);
  conv_epilogue<<<blks((size_t)NA * HID, TPB), TPB, 0, stream>>>(agg_a, rs_in_a, b_a2, xw_a, h16_a, NA, HID);

  // ---- conv2 (F): h_f2 in xw_f ----
  gemm_wmma<false, true, true, false><<<gemm_grid(NF, HID), 128, 0, stream>>>(
      dec16_f, wt_f2, nullptr, rs_out_f, xw_f, nullptr, NF, HID, HID);
  fill_f32<<<blks((size_t)NF * HID, TPB), TPB, 0, stream>>>(agg_f, 0.0f, NF * HID);
  edge_scatter<<<blks((size_t)EF * 32, TPB), TPB, 0, stream>>>(fcg_src, fcg_dst, xw_f, agg_f, EF);
  conv_epilogue<<<blks((size_t)NF * HID, TPB), TPB, 0, stream>>>(agg_f, rs_in_f, b_f2, xw_f, h16_f, NF, HID);

  // ---- graph readouts: mean over A nodes, max over F nodes ----
  fill_f32<<<1, HID, 0, stream>>>(col_a, 0.0f, HID);
  fill_f32<<<1, HID, 0, stream>>>(col_f, 0.0f, HID);   // int bits 0 == 0.0f; h_f2 >= 0
  colsum_kernel<<<512, HID, 0, stream>>>(xw_a, NA, HID, col_a);
  colmax_kernel<<<512, HID, 0, stream>>>(xw_f, NF, HID, (int*)col_f);

  // ---- normalize + (degenerate) attention + classifier ----
  finalize_kernel<<<1, HID, 0, stream>>>(col_a, col_f, W_cls, b_cls, (float*)d_out);
}
